// MNISTNet_34316788695269
// MI455X (gfx1250) — matrix operand order repaired and checked by reading, not compiled, not run on hardware
//
#include <hip/hip_runtime.h>

#define BATCH   65536
#define IN1     784
#define HD      128
#define GRID_N  10
#define KOUT    64
#define OUTD    10
#define KFLAT   (HD * GRID_N)   // 1280

#define WAVES       4
#define ROWS_PER_WG (WAVES * 16)
#define HSTRIDE     132          // padded fp32 stride for h tile (breaks bank conflicts)
#define YSTRIDE     68           // padded fp32 stride for y tile (16B aligned rows)
#define P1PLANE     (HD * 24)    // ushort plane stride for phase-1 staging [khalf][row][24]
#define P2PLANE     (KOUT * 24)  // ushort plane stride for phase-2 staging

typedef __bf16          v16bf __attribute__((ext_vector_type(16)));
typedef unsigned short  v16us __attribute__((ext_vector_type(16)));
typedef unsigned short  v8us  __attribute__((ext_vector_type(8)));
typedef float           v8f   __attribute__((ext_vector_type(8)));

// Round-to-nearest-even fp32 -> bf16 hi, residual -> bf16 lo (bf16x3 split).
__device__ __forceinline__ void bf16_split(float f, unsigned short& hi, unsigned short& lo) {
    unsigned int x  = __float_as_uint(f);
    unsigned int rh = x + 0x7FFFu + ((x >> 16) & 1u);
    hi = (unsigned short)(rh >> 16);
    float fh = __uint_as_float(((unsigned int)hi) << 16);
    float r  = f - fh;
    unsigned int y  = __float_as_uint(r);
    unsigned int rl = y + 0x7FFFu + ((y >> 16) & 1u);
    lo = (unsigned short)(rl >> 16);
}

__device__ __forceinline__ v8f wmma_bf16(v16us a, v16us b, v8f c) {
    return __builtin_amdgcn_wmma_f32_16x16x32_bf16(
        false, __builtin_bit_cast(v16bf, a),
        false, __builtin_bit_cast(v16bf, b),
        (short)0, c, false, false);
}

#define SHUF16(p0, p1) __builtin_shufflevector((p0), (p1), \
        0, 1, 2, 3, 4, 5, 6, 7, 8, 9, 10, 11, 12, 13, 14, 15)

__global__ __launch_bounds__(128)
void kan_fused_kernel(const float* __restrict__ x,
                      const float* __restrict__ w1,
                      const float* __restrict__ b1,
                      const float* __restrict__ coef,
                      const float* __restrict__ wbias,
                      const float* __restrict__ w2,
                      const float* __restrict__ b2,
                      float* __restrict__ out) {
    // Staging for w1 / coef chunks, pre-converted to bf16 hi/lo.
    // Layout: [khalf][row][24] ushorts; row stride 48B -> 16B-aligned v8us loads,
    // and 12*lane mod 64 is distinct for 16 lanes -> conflict-free fragment reads.
    __shared__ unsigned short stg_h[2 * HD * 24];
    __shared__ unsigned short stg_l[2 * HD * 24];
    __shared__ float lds_h[WAVES * 16 * HSTRIDE];   // per-wave h tiles; reused for y

    const int tid  = threadIdx.x;
    const int lane = tid & 31;
    const int wv   = tid >> 5;
    const int lm   = lane & 15;      // M (A row) or N (B/C col) within tile
    const int hs   = lane >> 4;      // lane-half select per ISA fragment layouts
    const int rowBase = blockIdx.x * ROWS_PER_WG + wv * 16;

    const v8f vzero = {0.f, 0.f, 0.f, 0.f, 0.f, 0.f, 0.f, 0.f};

    // ---------------- Phase 1: h = x @ w1.T + b1  (M=16/wave, N=128, K=784) ----
    v8f acc[8];
    #pragma unroll
    for (int nt = 0; nt < 8; ++nt) acc[nt] = vzero;

    const float* xrow = x + (size_t)(rowBase + lm) * IN1;

    for (int kc = 0; kc < IN1; kc += 32) {
        __syncthreads();
        // Cooperative stage of w1[:, kc:kc+32] as bf16 hi/lo (zero-pad K tail).
        for (int t = tid; t < HD * 32; t += 128) {
            int r = t >> 5, kk = t & 31, k = kc + kk;
            float f = (k < IN1) ? w1[r * IN1 + k] : 0.f;
            unsigned short h, l; bf16_split(f, h, l);
            int si = ((kk >> 3) & 1) * P1PLANE + r * 24 + ((kk >> 4) << 3) + (kk & 7);
            stg_h[si] = h; stg_l[si] = l;
        }
        __syncthreads();

        // A fragment (ISA 16-bit A 16x32): lanes<16 take K {0..7,16..23},
        // lanes>=16 take K {8..15,24..31} -> kb = kc + 8*hs.
        const int kb = kc + (hs << 3);
        float4 a0 = *(const float4*)(xrow + kb);
        float4 a1 = *(const float4*)(xrow + kb + 4);
        float4 a2, a3;
        if (kb + 16 < IN1) {
            a2 = *(const float4*)(xrow + kb + 16);
            a3 = *(const float4*)(xrow + kb + 20);
        } else {
            a2 = make_float4(0.f, 0.f, 0.f, 0.f);
            a3 = a2;
        }
        if (kc + 64 < IN1) __builtin_prefetch(xrow + kc + 64, 0, 0);

        float av[16] = {a0.x, a0.y, a0.z, a0.w, a1.x, a1.y, a1.z, a1.w,
                        a2.x, a2.y, a2.z, a2.w, a3.x, a3.y, a3.z, a3.w};
        v16us ah, al;
        #pragma unroll
        for (int j = 0; j < 16; ++j) {
            unsigned short h, l; bf16_split(av[j], h, l);
            ah[j] = h; al[j] = l;
        }

        #pragma unroll
        for (int nt = 0; nt < 8; ++nt) {
            const v8us* ph = (const v8us*)&stg_h[hs * P1PLANE + (nt * 16 + lm) * 24];
            const v8us* pl = (const v8us*)&stg_l[hs * P1PLANE + (nt * 16 + lm) * 24];
            v16us bh = SHUF16(ph[0], ph[1]);
            v16us bl = SHUF16(pl[0], pl[1]);
            // bf16x3: hi*hi + hi*lo + lo*hi ~= fp32 product
            acc[nt] = wmma_bf16(ah, bh, acc[nt]);
            acc[nt] = wmma_bf16(ah, bl, acc[nt]);
            acc[nt] = wmma_bf16(al, bh, acc[nt]);
        }
    }

    // h (+b1) -> LDS, padded stride, for transposed phase-2 reads.
    float* hbase = lds_h + wv * (16 * HSTRIDE);
    #pragma unroll
    for (int nt = 0; nt < 8; ++nt) {
        int col = nt * 16 + lm;
        float bias = b1[col];
        #pragma unroll
        for (int r = 0; r < 8; ++r)
            hbase[(hs * 8 + r) * HSTRIDE + col] = acc[nt][r] + bias;
    }

    // ---------------- Phase 2: y = einsum(psi(h), coef)  (K=1280, N=64) -------
    v8f yacc[4];
    #pragma unroll
    for (int ot = 0; ot < 4; ++ot) yacc[ot] = vzero;

    const float* hrow2 = hbase + lm * HSTRIDE;

    for (int kc = 0; kc < KFLAT; kc += 32) {
        __syncthreads();
        // Cooperative stage of coef[:, kc:kc+32] (flattened (i,g)) as bf16 hi/lo.
        for (int t = tid; t < KOUT * 32; t += 128) {
            int o = t >> 5, kk = t & 31;
            float f = coef[o * KFLAT + kc + kk];
            unsigned short h, l; bf16_split(f, h, l);
            int si = ((kk >> 3) & 1) * P2PLANE + o * 24 + ((kk >> 4) << 3) + (kk & 7);
            stg_h[si] = h; stg_l[si] = l;
        }
        __syncthreads();

        // A fragment: psi computed on the fly from the h tile in LDS.
        const int kb = kc + (hs << 3);
        v16us ah, al;
        #pragma unroll
        for (int j = 0; j < 16; ++j) {
            int idx = kb + ((j < 8) ? j : (8 + j));      // K per ISA A layout
            int i = idx / GRID_N;
            int g = idx - i * GRID_N;
            float tv = hrow2[i] * (float)(g + 1);
            float tt = tv * tv;
            // Ricker: C * (1 - t^2) * exp(-t^2/2)
            float p = 0.8673250705840776f * (1.f - tt) * __expf(-0.5f * tt);
            unsigned short h, l; bf16_split(p, h, l);
            ah[j] = h; al[j] = l;
        }

        #pragma unroll
        for (int ot = 0; ot < 4; ++ot) {
            const v8us* ph = (const v8us*)&stg_h[hs * P2PLANE + (ot * 16 + lm) * 24];
            const v8us* pl = (const v8us*)&stg_l[hs * P2PLANE + (ot * 16 + lm) * 24];
            v16us bh = SHUF16(ph[0], ph[1]);
            v16us bl = SHUF16(pl[0], pl[1]);
            yacc[ot] = wmma_bf16(ah, bh, yacc[ot]);
            yacc[ot] = wmma_bf16(ah, bl, yacc[ot]);
            yacc[ot] = wmma_bf16(al, bh, yacc[ot]);
        }
    }

    // y (+wbias) -> LDS (reuse this wave's h region; wave-local, LDS is in-order).
    float* yrow = hbase;
    #pragma unroll
    for (int ot = 0; ot < 4; ++ot) {
        float wb = wbias[ot * 16 + lm];
        #pragma unroll
        for (int r = 0; r < 8; ++r)
            yrow[(hs * 8 + r) * YSTRIDE + ot * 16 + lm] = yacc[ot][r] + wb;
    }

    // ---------------- Phase 3: out = y @ w2.T + b2  (tiny: VALU) --------------
    #pragma unroll
    for (int it = 0; it < 5; ++it) {
        int t  = lane + it * 32;            // 160 = 16 rows * 10 outputs per wave
        int r  = t / OUTD;
        int oo = t - r * OUTD;
        float s = b2[oo];
        const float* w2r = w2 + oo * KOUT;
        const float* yr  = yrow + r * YSTRIDE;
        #pragma unroll 8
        for (int o = 0; o < KOUT; ++o) s += yr[o] * w2r[o];
        out[(size_t)(rowBase + r) * OUTD + oo] = s;
    }
}

extern "C" void kernel_launch(void* const* d_in, const int* in_sizes, int n_in,
                              void* d_out, int out_size, void* d_ws, size_t ws_size,
                              hipStream_t stream) {
    (void)in_sizes; (void)n_in; (void)out_size; (void)d_ws; (void)ws_size;
    const float* x     = (const float*)d_in[0];
    const float* w1    = (const float*)d_in[1];
    const float* b1    = (const float*)d_in[2];
    const float* coef  = (const float*)d_in[3];
    const float* wbias = (const float*)d_in[4];
    const float* w2    = (const float*)d_in[5];
    const float* b2    = (const float*)d_in[6];
    float* out = (float*)d_out;

    dim3 grid(BATCH / ROWS_PER_WG);   // 1024 workgroups
    dim3 block(128);                  // 4 waves (wave32)
    kan_fused_kernel<<<grid, block, 0, stream>>>(x, w1, b1, coef, wbias, w2, b2, out);
}
